// ExtractorModel_68753836474858
// MI455X (gfx1250) — matrix operand order repaired and checked by reading, not compiled, not run on hardware
//
#include <hip/hip_runtime.h>
#include <hip/hip_bf16.h>

typedef __attribute__((ext_vector_type(16))) _Float16 v16h;
typedef __attribute__((ext_vector_type(8)))  _Float16 v8h;
typedef __attribute__((ext_vector_type(8)))  float    v8f;

#define T_SEQ  128
#define BATCH  512
#define D_IN   768
#define M_TOT  (T_SEQ * BATCH)   // 65536 rows

__device__ __forceinline__ float sigmoidf_(float x) {
    return 1.0f / (1.0f + __expf(-x));
}
__device__ __forceinline__ float tanhf_(float x) {
    x = fminf(fmaxf(x, -15.0f), 15.0f);
    float e = __expf(2.0f * x);
    return (e - 1.0f) / (e + 1.0f);
}

// ---------------------------------------------------------------------------
// K0: one-time prep. Convert combined bi-LSTM input weights [Wih_f; Wih_b]
// (64,768) to f16 in workspace; precompute combined biases (64).
// ---------------------------------------------------------------------------
__global__ __launch_bounds__(256)
void k0_prep(const float* __restrict__ Wf, const float* __restrict__ Wb,
             const float* __restrict__ bihf, const float* __restrict__ bhhf,
             const float* __restrict__ bihb, const float* __restrict__ bhhb,
             _Float16* __restrict__ wc16, float* __restrict__ bias_c)
{
    const int i = blockIdx.x * 256 + threadIdx.x;
    if (i < 64 * D_IN) {
        const int n = i / D_IN, k = i - n * D_IN;
        const float v = (n < 32) ? Wf[n * D_IN + k] : Wb[(n - 32) * D_IN + k];
        wc16[i] = (_Float16)v;
    }
    if (i < 32)       bias_c[i] = bihf[i] + bhhf[i];
    else if (i < 64)  bias_c[i] = bihb[i - 32] + bhhb[i - 32];
}

// ---------------------------------------------------------------------------
// K1: bi-LSTM input projection GEMM, M=65536 K=768 N=64.
// One wave owns a 16-row M-tile and ALL 4 N-tiles (4 f32 accumulators), so the
// 201 MB X tensor is streamed exactly once (bandwidth floor ~9 us @ 23.3 TB/s).
// B fragments come from the f16-preconverted weight copy (L2-resident).
// Inner loop per wave: 4x b128 A-loads (+cvt), 8x b128 f16 B-loads, 4x WMMA.
// ---------------------------------------------------------------------------
__global__ __launch_bounds__(256)
void k1_bilstm_inproj(const float* __restrict__ X,
                      const _Float16* __restrict__ wc16,  // (64,768) f16
                      const float* __restrict__ bias_c,   // (64)
                      float* __restrict__ G)              // (65536,64)
{
    const int lane = threadIdx.x & 31;
    const int wave = threadIdx.x >> 5;                // 8 waves = 8 M-tiles
    const int m0 = blockIdx.x * 128 + wave * 16;
    const int hi = lane >> 4;                         // half-wave K group
    const int row = m0 + (lane & 15);                 // A: lane%16 = M row
    const int nb = lane & 15;                         // B: lane%16 = N within tile

    v8f acc0 = {}, acc1 = {}, acc2 = {}, acc3 = {};
    const float* arow = X + (size_t)row * D_IN;

    for (int k0 = 0; k0 < D_IN; k0 += 32) {
        // ---- A fragment (16x32 f16): halves cover K = kb..kb+7, kb+16..kb+23,
        //      kb = k0 + 8*(lane>=16)
        const float* ap = arow + k0 + (hi << 3);
        __builtin_prefetch(ap + 32, 0, 1);            // next K block
        float4 a0 = *(const float4*)(ap + 0);
        float4 a1 = *(const float4*)(ap + 4);
        float4 a2 = *(const float4*)(ap + 16);
        float4 a3 = *(const float4*)(ap + 20);
        v16h af;
        af[0]=(_Float16)a0.x; af[1]=(_Float16)a0.y; af[2]=(_Float16)a0.z; af[3]=(_Float16)a0.w;
        af[4]=(_Float16)a1.x; af[5]=(_Float16)a1.y; af[6]=(_Float16)a1.z; af[7]=(_Float16)a1.w;
        af[8]=(_Float16)a2.x; af[9]=(_Float16)a2.y; af[10]=(_Float16)a2.z; af[11]=(_Float16)a2.w;
        af[12]=(_Float16)a3.x; af[13]=(_Float16)a3.y; af[14]=(_Float16)a3.z; af[15]=(_Float16)a3.w;

        // ---- B fragments for the 4 N-tiles: lane = N col, halves cover
        //      K = kb..kb+7, kb+16..kb+23, kb = k0 + 8*(lane>=16); straight f16 loads.
        const _Float16* bbase = wc16 + (size_t)nb * D_IN + k0 + (hi << 3);
        #define LOAD_B(NT) \
            __builtin_shufflevector(*(const v8h*)(bbase + (NT) * 16 * D_IN), \
                                    *(const v8h*)(bbase + (NT) * 16 * D_IN + 16), \
                                    0,1,2,3,4,5,6,7,8,9,10,11,12,13,14,15)
        v16h bf0 = LOAD_B(0);
        v16h bf1 = LOAD_B(1);
        v16h bf2 = LOAD_B(2);
        v16h bf3 = LOAD_B(3);
        #undef LOAD_B

        acc0 = __builtin_amdgcn_wmma_f32_16x16x32_f16(false, af, false, bf0, (short)0, acc0, false, false);
        acc1 = __builtin_amdgcn_wmma_f32_16x16x32_f16(false, af, false, bf1, (short)0, acc1, false, false);
        acc2 = __builtin_amdgcn_wmma_f32_16x16x32_f16(false, af, false, bf2, (short)0, acc2, false, false);
        acc3 = __builtin_amdgcn_wmma_f32_16x16x32_f16(false, af, false, bf3, (short)0, acc3, false, false);
    }

    // C/D layout: VGPR r -> M = r + 8*halfwave, N = lane%16
    #pragma unroll
    for (int nt = 0; nt < 4; ++nt) {
        const int col = nt * 16 + nb;
        const float bias = bias_c[col];
        const v8f acc = (nt == 0) ? acc0 : (nt == 1) ? acc1 : (nt == 2) ? acc2 : acc3;
        #pragma unroll
        for (int r = 0; r < 8; ++r) {
            const int orow = m0 + r + (hi << 3);
            G[(size_t)orow * 64 + col] = acc[r] + bias;
        }
    }
}

// ---------------------------------------------------------------------------
// K2: bi-LSTM recurrence. One thread per (batch, direction); independent scans.
// ---------------------------------------------------------------------------
__global__ __launch_bounds__(256)
void k2_bilstm_scan(const float* __restrict__ G,       // (65536,64)
                    const float* __restrict__ Whh_f,   // (32,8)
                    const float* __restrict__ Whh_b,   // (32,8)
                    float* __restrict__ hcat)          // (T,B,16)
{
    __shared__ float sW[2][32 * 8];
    for (int i = threadIdx.x; i < 256; i += blockDim.x) {
        sW[0][i] = Whh_f[i];
        sW[1][i] = Whh_b[i];
    }
    __syncthreads();

    const int tid = blockIdx.x * blockDim.x + threadIdx.x;
    if (tid >= 2 * BATCH) return;
    const int b = tid & (BATCH - 1);
    const int dir = tid >> 9;                 // 0=fwd, 1=bwd
    const float* W = sW[dir];

    float h[8], c[8];
    #pragma unroll
    for (int k = 0; k < 8; ++k) { h[k] = 0.0f; c[k] = 0.0f; }

    for (int s = 0; s < T_SEQ; ++s) {
        const int t = dir ? (T_SEQ - 1 - s) : s;
        const float* g = G + ((size_t)t * BATCH + b) * 64 + dir * 32;
        float hn[8];
        #pragma unroll
        for (int k = 0; k < 8; ++k) {
            float gi = g[k], gf = g[8 + k], gg = g[16 + k], go = g[24 + k];
            #pragma unroll
            for (int j = 0; j < 8; ++j) {
                const float hj = h[j];
                gi += W[k * 8 + j] * hj;
                gf += W[(8 + k) * 8 + j] * hj;
                gg += W[(16 + k) * 8 + j] * hj;
                go += W[(24 + k) * 8 + j] * hj;
            }
            const float i_ = sigmoidf_(gi), f_ = sigmoidf_(gf), o_ = sigmoidf_(go);
            c[k] = f_ * c[k] + i_ * tanhf_(gg);
            hn[k] = o_ * tanhf_(c[k]);
        }
        float* hp = hcat + ((size_t)t * BATCH + b) * 16 + dir * 8;
        #pragma unroll
        for (int k = 0; k < 8; ++k) { h[k] = hn[k]; hp[k] = hn[k]; }
    }
}

// ---------------------------------------------------------------------------
// K3: ptr-LSTM input projection. M=65536, K=16 (padded to 32), N=64. 1 WMMA/wave.
// ---------------------------------------------------------------------------
__global__ __launch_bounds__(256)
void k3_ptr_inproj(const float* __restrict__ hcat,   // (65536,16)
                   const float* __restrict__ Wp,     // (64,16)
                   const float* __restrict__ bihp, const float* __restrict__ bhhp,
                   float* __restrict__ Gp)           // (65536,64)
{
    const int lane = threadIdx.x & 31;
    const int wave = threadIdx.x >> 5;
    const int mt = wave >> 2;
    const int nt = wave & 3;
    const int m0 = blockIdx.x * 32 + mt * 16;
    const int hi = lane >> 4;
    const int row = m0 + (lane & 15);
    const int n   = nt * 16 + (lane & 15);

    v16h af, bf;
    #pragma unroll
    for (int i = 0; i < 16; ++i) { af[i] = (_Float16)0.0f; bf[i] = (_Float16)0.0f; }

    // A: valid K = 0..15 -> halves 0..7 of each half-wave (kb = 8*hi)
    const float* ap = hcat + (size_t)row * 16 + (hi << 3);
    float4 a0 = *(const float4*)(ap + 0);
    float4 a1 = *(const float4*)(ap + 4);
    af[0]=(_Float16)a0.x; af[1]=(_Float16)a0.y; af[2]=(_Float16)a0.z; af[3]=(_Float16)a0.w;
    af[4]=(_Float16)a1.x; af[5]=(_Float16)a1.y; af[6]=(_Float16)a1.z; af[7]=(_Float16)a1.w;

    // B: valid K = 0..15 -> elements 0..7 of each half-wave (kb = 8*hi), rest zero pad
    {
        const float* bp = Wp + (size_t)n * 16 + (hi << 3);
        float4 b0 = *(const float4*)(bp + 0);
        float4 b1 = *(const float4*)(bp + 4);
        bf[0]=(_Float16)b0.x; bf[1]=(_Float16)b0.y; bf[2]=(_Float16)b0.z; bf[3]=(_Float16)b0.w;
        bf[4]=(_Float16)b1.x; bf[5]=(_Float16)b1.y; bf[6]=(_Float16)b1.z; bf[7]=(_Float16)b1.w;
    }

    v8f acc = {};
    acc = __builtin_amdgcn_wmma_f32_16x16x32_f16(false, af, false, bf,
                                                 (short)0, acc, false, false);
    const float bias = bihp[n] + bhhp[n];
    #pragma unroll
    for (int r = 0; r < 8; ++r) {
        const int orow = m0 + r + (hi << 3);
        Gp[(size_t)orow * 64 + n] = acc[r] + bias;
    }
}

// ---------------------------------------------------------------------------
// K4: ptr-LSTM recurrence. 4 threads per batch element (each owns 4 hidden
// units); hidden state shared via double-buffered LDS, one barrier per step.
// 64 batch elements per 256-thread block.
// ---------------------------------------------------------------------------
__global__ __launch_bounds__(256)
void k4_ptr_scan(const float* __restrict__ Gp,      // (65536,64)
                 const float* __restrict__ Whh_p,   // (64,16)
                 float* __restrict__ z)             // (T,B,16)
{
    __shared__ float sW[64 * 16];
    __shared__ float hbuf[2][64][16];
    for (int i = threadIdx.x; i < 1024; i += 256) sW[i] = Whh_p[i];

    const int lb = threadIdx.x >> 2;   // local batch 0..63
    const int q  = threadIdx.x & 3;    // quarter: hidden units 4q..4q+3
    const int b  = blockIdx.x * 64 + lb;
    #pragma unroll
    for (int u = 0; u < 4; ++u) hbuf[0][lb][q * 4 + u] = 0.0f;
    __syncthreads();

    float c[4] = {0.0f, 0.0f, 0.0f, 0.0f};
    int p = 0;
    for (int t = 0; t < T_SEQ; ++t) {
        const float* hprev = hbuf[p][lb];
        const float* g = Gp + ((size_t)t * BATCH + b) * 64;
        float hn[4];
        #pragma unroll
        for (int u = 0; u < 4; ++u) {
            const int k = q * 4 + u;
            float gi = g[k], gf = g[16 + k], gg = g[32 + k], go = g[48 + k];
            #pragma unroll
            for (int j = 0; j < 16; ++j) {
                const float hj = hprev[j];
                gi += sW[k * 16 + j] * hj;
                gf += sW[(16 + k) * 16 + j] * hj;
                gg += sW[(32 + k) * 16 + j] * hj;
                go += sW[(48 + k) * 16 + j] * hj;
            }
            const float i_ = sigmoidf_(gi), f_ = sigmoidf_(gf), o_ = sigmoidf_(go);
            c[u] = f_ * c[u] + i_ * tanhf_(gg);
            hn[u] = o_ * tanhf_(c[u]);
        }
        float* zt = z + ((size_t)t * BATCH + b) * 16 + q * 4;
        #pragma unroll
        for (int u = 0; u < 4; ++u) {
            hbuf[p ^ 1][lb][q * 4 + u] = hn[u];
            zt[u] = hn[u];
        }
        __syncthreads();
        p ^= 1;
    }
}

// ---------------------------------------------------------------------------
// K5: per-timestep attention + output head. One block per t, 512 threads.
// attn[b,c] = h[b].z[c]; softmax over b (per column c, online max/sum);
// ctx[b] = sum_c w[b,c] h[c]; p = sigmoid(W_v . tanh(W_h h + W_e ctx)).
// ---------------------------------------------------------------------------
__global__ __launch_bounds__(512)
void k5_attn(const float* __restrict__ hcat,  // (T,B,16)
             const float* __restrict__ zbuf,  // (T,B,16)
             const float* __restrict__ W_h,   // (16,16)
             const float* __restrict__ W_e,   // (16,16)
             const float* __restrict__ W_v,   // (1,16)
             float* __restrict__ out)         // (T,B)
{
    __shared__ float sh[BATCH][16];   // 32 KB
    __shared__ float sM[BATCH];
    __shared__ float sR[BATCH];       // 1/sum
    __shared__ float sWh[256], sWe[256], sWv[16];

    const int t = blockIdx.x;
    const int tid = threadIdx.x;
    const float* ht = hcat + (size_t)t * BATCH * 16;
    const float* zt = zbuf + (size_t)t * BATCH * 16;

    {   // stage h[t] tile
        const float4* hp = (const float4*)(ht + (size_t)tid * 16);
        float4* dst = (float4*)sh[tid];
        dst[0] = hp[0]; dst[1] = hp[1]; dst[2] = hp[2]; dst[3] = hp[3];
    }
    if (tid < 256) { sWh[tid] = W_h[tid]; sWe[tid] = W_e[tid]; }
    if (tid < 16)  { sWv[tid] = W_v[tid]; }
    __syncthreads();

    // Pass 1: thread = column c. Online softmax stats over b.
    {
        float zc[16];
        const float4* zp = (const float4*)(zt + (size_t)tid * 16);
        float4 z0 = zp[0], z1 = zp[1], z2 = zp[2], z3 = zp[3];
        zc[0]=z0.x; zc[1]=z0.y; zc[2]=z0.z; zc[3]=z0.w;
        zc[4]=z1.x; zc[5]=z1.y; zc[6]=z1.z; zc[7]=z1.w;
        zc[8]=z2.x; zc[9]=z2.y; zc[10]=z2.z; zc[11]=z2.w;
        zc[12]=z3.x; zc[13]=z3.y; zc[14]=z3.z; zc[15]=z3.w;

        float m = -3.4e38f, S = 0.0f;
        for (int b = 0; b < BATCH; ++b) {
            float s = 0.0f;
            #pragma unroll
            for (int d = 0; d < 16; ++d) s += sh[b][d] * zc[d];
            if (s > m) { S = S * __expf(m - s) + 1.0f; m = s; }
            else       { S += __expf(s - m); }
        }
        sM[tid] = m;
        sR[tid] = 1.0f / S;
    }
    __syncthreads();

    // Pass 2: thread = row b. Recompute scores, accumulate context.
    float hb[16];
    #pragma unroll
    for (int d = 0; d < 16; ++d) hb[d] = sh[tid][d];
    float ctx[16];
    #pragma unroll
    for (int d = 0; d < 16; ++d) ctx[d] = 0.0f;

    for (int c = 0; c < BATCH; ++c) {
        const float* zc = zt + (size_t)c * 16;   // uniform address -> scalar-cached
        float s = 0.0f;
        #pragma unroll
        for (int d = 0; d < 16; ++d) s += hb[d] * zc[d];
        const float w = __expf(s - sM[c]) * sR[c];
        #pragma unroll
        for (int d = 0; d < 16; ++d) ctx[d] += w * sh[c][d];
    }

    float pacc = 0.0f;
    #pragma unroll
    for (int d = 0; d < 16; ++d) {
        float u = 0.0f;
        #pragma unroll
        for (int e = 0; e < 16; ++e)
            u += sWh[d * 16 + e] * hb[e] + sWe[d * 16 + e] * ctx[e];
        pacc += sWv[d] * tanhf_(u);
    }
    out[(size_t)t * BATCH + tid] = sigmoidf_(pacc);
}

// ---------------------------------------------------------------------------
extern "C" void kernel_launch(void* const* d_in, const int* in_sizes, int n_in,
                              void* d_out, int out_size, void* d_ws, size_t ws_size,
                              hipStream_t stream) {
    const float* X     = (const float*)d_in[0];
    const float* Wih_f = (const float*)d_in[1];
    const float* Whh_f = (const float*)d_in[2];
    const float* bih_f = (const float*)d_in[3];
    const float* bhh_f = (const float*)d_in[4];
    const float* Wih_b = (const float*)d_in[5];
    const float* Whh_b = (const float*)d_in[6];
    const float* bih_b = (const float*)d_in[7];
    const float* bhh_b = (const float*)d_in[8];
    const float* Wih_p = (const float*)d_in[9];
    const float* Whh_p = (const float*)d_in[10];
    const float* bih_p = (const float*)d_in[11];
    const float* bhh_p = (const float*)d_in[12];
    const float* W_h   = (const float*)d_in[13];
    const float* W_e   = (const float*)d_in[14];
    const float* W_v   = (const float*)d_in[15];
    float* out = (float*)d_out;

    char* ws = (char*)d_ws;
    const size_t szG = (size_t)M_TOT * 64 * sizeof(float);   // 16.8 MB
    const size_t szH = (size_t)M_TOT * 16 * sizeof(float);   // 4.2 MB
    float*     G      = (float*)(ws);
    float*     Gp     = (float*)(ws + szG);
    float*     hc     = (float*)(ws + 2 * szG);
    float*     zz     = (float*)(ws + 2 * szG + szH);
    _Float16*  wc16   = (_Float16*)(ws + 2 * szG + 2 * szH);
    float*     bias_c = (float*)(ws + 2 * szG + 2 * szH + (size_t)64 * D_IN * sizeof(_Float16));

    k0_prep<<<(64 * D_IN + 255) / 256, 256, 0, stream>>>(Wih_f, Wih_b,
                                                         bih_f, bhh_f, bih_b, bhh_b,
                                                         wc16, bias_c);
    k1_bilstm_inproj<<<M_TOT / 128, 256, 0, stream>>>(X, wc16, bias_c, G);
    k2_bilstm_scan<<<4, 256, 0, stream>>>(G, Whh_f, Whh_b, hc);
    k3_ptr_inproj<<<M_TOT / 32, 256, 0, stream>>>(hc, Wih_p, bih_p, bhh_p, Gp);
    k4_ptr_scan<<<BATCH / 64, 256, 0, stream>>>(Gp, Whh_p, zz);
    k5_attn<<<T_SEQ, 512, 0, stream>>>(hc, zz, W_h, W_e, W_v, out);
}
